// SelfAttention_81217831567852
// MI455X (gfx1250) — matrix operand order repaired and checked by reading, not compiled, not run on hardware
//
#include <hip/hip_runtime.h>

// Problem constants (B, C, N) from the reference.
#define B_ 8
#define C_ 64
#define N_ 4096
#define BN 64   // key rows (n) per workgroup (4 waves x 16 rows)
#define BM 64   // query cols (m) per inner block
#define RS 80   // LDS row stride in u16 (64 data + 16 pad -> 160B, 16B aligned)

typedef __attribute__((ext_vector_type(16))) __bf16       v16bf;
typedef __attribute__((ext_vector_type(2)))  __bf16       v2bf;
typedef __attribute__((ext_vector_type(8)))  float        v8f;
typedef __attribute__((ext_vector_type(2)))  float        f32x2;
typedef __attribute__((ext_vector_type(4)))  float        f32x4;
typedef __attribute__((ext_vector_type(4)))  unsigned int u32x4;
typedef __attribute__((ext_vector_type(2)))  unsigned int u32x2;

union Frag { u32x4 u4[2]; v16bf bf; };   // 32 bytes = 8 VGPRs = one WMMA A/B operand

// ---- fp32x2 -> packed bf16x2 (RNE). Backend selects v_cvt_pk_bf16_f32 when present.
__device__ inline unsigned pk_bf16(float x, float y) {
#if defined(__HIP_DEVICE_COMPILE__) && __has_builtin(__builtin_amdgcn_cvt_pk_bf16_f32)
  v2bf r = __builtin_amdgcn_cvt_pk_bf16_f32(x, y);
  return __builtin_bit_cast(unsigned, r);
#else
  f32x2 in = {x, y};
  v2bf r = __builtin_convertvector(in, v2bf);   // fptrunc, RNE
  return __builtin_bit_cast(unsigned, r);
#endif
}
__device__ inline unsigned short f2bf(float f) {
  __bf16 h = (__bf16)f;
  return __builtin_bit_cast(unsigned short, h);
}

// ---- raw hardware exp2 (v_exp_f32 is base-2): 1 VALU per call, no fixup.
__device__ inline float fast_exp2(float x) {
#if defined(__HIP_DEVICE_COMPILE__) && __has_builtin(__builtin_amdgcn_exp2f)
  return __builtin_amdgcn_exp2f(x);
#else
  return exp2f(x);
#endif
}

// Butterfly reductions over 16-lane halves (wave32: masks 1,2,4,8 stay in-half).
__device__ inline float red_max16(float x) {
  x = fmaxf(x, __shfl_xor(x, 1, 32));
  x = fmaxf(x, __shfl_xor(x, 2, 32));
  x = fmaxf(x, __shfl_xor(x, 4, 32));
  x = fmaxf(x, __shfl_xor(x, 8, 32));
  return x;
}
__device__ inline float red_sum16(float x) {
  x += __shfl_xor(x, 1, 32);
  x += __shfl_xor(x, 2, 32);
  x += __shfl_xor(x, 4, 32);
  x += __shfl_xor(x, 8, 32);
  return x;
}

__global__ __launch_bounds__(128, 1)
void attn_fwd_flash(const float* __restrict__ K,
                    const float* __restrict__ Q,
                    const float* __restrict__ V,
                    float* __restrict__ Out) {
  // LDS staging: all tiles bf16, padded rows for aligned ds_load_b128.
  __shared__ __align__(16) unsigned short lds_k[BN * RS]; // [n_local][c]
  __shared__ __align__(16) unsigned short lds_q[BM * RS]; // [m_local][c]
  __shared__ __align__(16) unsigned short lds_v[C_ * RS]; // [c][m_local]
  __shared__ __align__(16) unsigned short lds_p[BN * RS]; // per-wave P scratch [n_local][m_local]

  const int b    = blockIdx.x / (N_ / BN);
  const int n0   = (blockIdx.x % (N_ / BN)) * BN;
  const int tid  = threadIdx.x;
  const int w    = tid >> 5;       // wave id 0..3
  const int lane = tid & 31;
  const int l16  = lane & 15;
  const int hi   = lane >> 4;      // lane half: 0 or 1
  // Fold softmax scale AND log2(e) into K at staging time: scores land directly
  // in the exp2 domain, so softmax needs zero extra multiplies (v_exp_f32 is base-2).
  const float kscale = 0.18033688f;  // (1/sqrt(64)) * log2(e)

  const float* Kb = K + (size_t)b * C_ * N_;
  const float* Qb = Q + (size_t)b * C_ * N_;
  const float* Vb = V + (size_t)b * C_ * N_;
  float*       Ob = Out + (size_t)b * C_ * N_;

  // ---- Stage K tile once: global k[c][n] (coalesced over n) -> LDS [n][c] bf16
  for (int it = 0; it < 8; ++it) {
    int lin = it * 128 + tid;          // 0..1023 covers 64c x 16 n-quads
    int c   = lin >> 4;
    int n4  = (lin & 15) << 2;
    f32x4 kv = *(const f32x4*)(Kb + (size_t)c * N_ + n0 + n4);
    unsigned p01 = pk_bf16(kv.x * kscale, kv.y * kscale);
    unsigned p23 = pk_bf16(kv.z * kscale, kv.w * kscale);
    lds_k[(n4 + 0) * RS + c] = (unsigned short)p01;
    lds_k[(n4 + 1) * RS + c] = (unsigned short)(p01 >> 16);
    lds_k[(n4 + 2) * RS + c] = (unsigned short)p23;
    lds_k[(n4 + 3) * RS + c] = (unsigned short)(p23 >> 16);
  }
  __syncthreads();

  // ---- Build K A-fragments (held in registers for the whole kernel).
  // A-layout (16-bit, 16x32): VGPR0-7 = K(hi*16 + 0..15), matching the B fragments
  Frag ak[2];
  {
    const unsigned short* pr = &lds_k[(w * 16 + l16) * RS];
#pragma unroll
    for (int f = 0; f < 2; ++f) {
      ak[f].u4[0] = *(const u32x4*)(pr + f * 32 + hi * 16);
      ak[f].u4[1] = *(const u32x4*)(pr + f * 32 + hi * 16 + 8);
    }
  }

  // ---- Online-softmax state (log2 domain). Row r of this lane = n0 + w*16 + 8*hi + r.
  float m_i[8], l_i[8];
  v8f acco[4] = {};   // O tile 16n x 64c: 4 c-subtiles in D-layout
#pragma unroll
  for (int r = 0; r < 8; ++r) { m_i[r] = -1e30f; l_i[r] = 0.0f; }

  for (int mb = 0; mb < N_; mb += BM) {
    __syncthreads();  // previous iteration's lds_q/lds_v readers are done

    // ---- Stage Q (transposed -> [m][c]) and V (-> [c][m]) tiles, bf16.
    for (int it = 0; it < 8; ++it) {
      int lin = it * 128 + tid;
      int c   = lin >> 4;
      int m4  = (lin & 15) << 2;
      const float* qs = Qb + (size_t)c * N_ + mb + m4;
      const float* vs = Vb + (size_t)c * N_ + mb + m4;
      if (mb + BM < N_) {              // prefetch next block -> global_prefetch_b8
        __builtin_prefetch(qs + BM, 0, 0);
        __builtin_prefetch(vs + BM, 0, 0);
      }
      f32x4 qv = *(const f32x4*)qs;
      unsigned q01 = pk_bf16(qv.x, qv.y);
      unsigned q23 = pk_bf16(qv.z, qv.w);
      lds_q[(m4 + 0) * RS + c] = (unsigned short)q01;
      lds_q[(m4 + 1) * RS + c] = (unsigned short)(q01 >> 16);
      lds_q[(m4 + 2) * RS + c] = (unsigned short)q23;
      lds_q[(m4 + 3) * RS + c] = (unsigned short)(q23 >> 16);
      f32x4 vv = *(const f32x4*)vs;
      u32x2 pk;
      pk.x = pk_bf16(vv.x, vv.y);
      pk.y = pk_bf16(vv.z, vv.w);
      *(u32x2*)&lds_v[c * RS + m4] = pk;
    }
    __syncthreads();

    // ---- Scores: S(16n x 64m) = K'tile * Qtile (already in log2 domain).
    // B-layout used here (16-bit, 32x16): col = l16, VGPR0-7 = K(hi*16 + 0..15), same k order as A
    float S[4][8];
#pragma unroll
    for (int t = 0; t < 4; ++t) {
      const unsigned short* qr = &lds_q[(t * 16 + l16) * RS];
      Frag bq0, bq1;
      bq0.u4[0] = *(const u32x4*)(qr + hi * 16 + 0);
      bq0.u4[1] = *(const u32x4*)(qr + hi * 16 + 8);
      bq1.u4[0] = *(const u32x4*)(qr + 32 + hi * 16 + 0);
      bq1.u4[1] = *(const u32x4*)(qr + 32 + hi * 16 + 8);
      v8f acc = {};
      acc = __builtin_amdgcn_wmma_f32_16x16x32_bf16(false, ak[0].bf, false, bq0.bf,
                                                    (short)0, acc, false, false);
      acc = __builtin_amdgcn_wmma_f32_16x16x32_bf16(false, ak[1].bf, false, bq1.bf,
                                                    (short)0, acc, false, false);
#pragma unroll
      for (int r = 0; r < 8; ++r) S[t][r] = acc[r];
    }

    // ---- Online softmax over the 64 m columns of this block (base-2).
    float mnew[8], alpha[8], rsum[8];
#pragma unroll
    for (int r = 0; r < 8; ++r) {
      float v = fmaxf(fmaxf(S[0][r], S[1][r]), fmaxf(S[2][r], S[3][r]));
      v = red_max16(v);
      mnew[r]  = fmaxf(m_i[r], v);
      alpha[r] = fast_exp2(m_i[r] - mnew[r]);   // single v_exp_f32
      rsum[r]  = 0.0f;
    }
#pragma unroll
    for (int t = 0; t < 4; ++t)
#pragma unroll
      for (int r = 0; r < 8; ++r) {
        float p = fast_exp2(S[t][r] - mnew[r]); // single v_exp_f32
        S[t][r] = p;
        rsum[r] += p;
      }
#pragma unroll
    for (int r = 0; r < 8; ++r) {
      l_i[r] = l_i[r] * alpha[r] + red_sum16(rsum[r]);
      m_i[r] = mnew[r];
    }

    // ---- P: D-layout -> row-major [n][m] bf16 via per-wave LDS scratch.
    {
      unsigned short* pw = &lds_p[w * 16 * RS];
#pragma unroll
      for (int t = 0; t < 4; ++t)
#pragma unroll
        for (int r = 0; r < 8; ++r)
          pw[(r + 8 * hi) * RS + t * 16 + l16] = f2bf(S[t][r]);
    }
    // Same-wave LDS RAW: wait for the stores before fragment reads.
    asm volatile("s_wait_dscnt 0" ::: "memory");

    // ---- Rescale running O by alpha (per-row).
#pragma unroll
    for (int t = 0; t < 4; ++t)
#pragma unroll
      for (int r = 0; r < 8; ++r)
        acco[t][r] *= alpha[r];

    // ---- P A-fragments from LDS scratch.
    Frag pa[2];
    {
      const unsigned short* pr = &lds_p[(w * 16 + l16) * RS];
#pragma unroll
      for (int f = 0; f < 2; ++f) {
        pa[f].u4[0] = *(const u32x4*)(pr + f * 32 + hi * 16);
        pa[f].u4[1] = *(const u32x4*)(pr + f * 32 + hi * 16 + 8);
      }
    }

    // ---- O(16n x 64c) += P(16n x 64m) * V^T(64m x 64c), 4 c-subtiles x 2 K-chunks.
#pragma unroll
    for (int t = 0; t < 4; ++t) {
      const unsigned short* vr = &lds_v[(t * 16 + l16) * RS];
      Frag bv0, bv1;
      bv0.u4[0] = *(const u32x4*)(vr + hi * 16 + 0);
      bv0.u4[1] = *(const u32x4*)(vr + hi * 16 + 8);
      bv1.u4[0] = *(const u32x4*)(vr + 32 + hi * 16 + 0);
      bv1.u4[1] = *(const u32x4*)(vr + 32 + hi * 16 + 8);
      acco[t] = __builtin_amdgcn_wmma_f32_16x16x32_bf16(false, pa[0].bf, false, bv0.bf,
                                                        (short)0, acco[t], false, false);
      acco[t] = __builtin_amdgcn_wmma_f32_16x16x32_bf16(false, pa[1].bf, false, bv1.bf,
                                                        (short)0, acco[t], false, false);
    }
  }

  // ---- Epilogue: out[b][c][n] = O[n][c] / l[n]
  float invl[8];
#pragma unroll
  for (int r = 0; r < 8; ++r) invl[r] = 1.0f / l_i[r];
  const int nbase = n0 + w * 16 + 8 * hi;
#pragma unroll
  for (int t = 0; t < 4; ++t) {
    int c = t * 16 + l16;
    float* op = Ob + (size_t)c * N_ + nbase;
#pragma unroll
    for (int r = 0; r < 8; ++r) op[r] = acco[t][r] * invl[r];
  }
}

extern "C" void kernel_launch(void* const* d_in, const int* in_sizes, int n_in,
                              void* d_out, int out_size, void* d_ws, size_t ws_size,
                              hipStream_t stream) {
  (void)in_sizes; (void)n_in; (void)out_size; (void)d_ws; (void)ws_size;
  const float* k = (const float*)d_in[0];
  const float* q = (const float*)d_in[1];
  const float* v = (const float*)d_in[2];
  float* out = (float*)d_out;
  dim3 grid(B_ * (N_ / BN));   // 8 batches x 64 n-blocks = 512 workgroups
  dim3 block(128);             // 4 waves; each wave owns a 16x64 output tile
  hipLaunchKernelGGL(attn_fwd_flash, grid, block, 0, stream, k, q, v, out);
}
